// BinaryTreeConv_77575699300801
// MI455X (gfx1250) — matrix operand order repaired and checked by reading, not compiled, not run on hardware
//
#include <hip/hip_runtime.h>
#include <hip/hip_bf16.h>

typedef __attribute__((ext_vector_type(16))) __bf16 v16bf;
typedef __attribute__((ext_vector_type(8)))  float  v8f;

#define BB   64
#define NN   2048
#define CC   256
#define KK   768      // 3*CC
#define OUTC 256
#define PADROWS 2049  // NN+1, row 0 is the zero pad row
#define OG   64       // outs per block (W slice in LDS)
#define NWAVE 8       // waves per block
#define TILES_PER_WAVE 4
#define BLKS_PER_OG 256   // 256 blocks * 8 waves * 4 tiles = 8192 node tiles

// round-to-nearest-even f32 -> bf16 (stored as raw ushort)
static __device__ __forceinline__ unsigned short f2bf(float f) {
    unsigned int u = __float_as_uint(f);
    u += 0x7FFFu + ((u >> 16) & 1u);
    return (unsigned short)(u >> 16);
}

// ---- prep: W [256][768] f32 -> bf16 -------------------------------------
__global__ void convert_w_kernel(const float* __restrict__ W,
                                 unsigned short* __restrict__ Wb, int n) {
    int i = blockIdx.x * blockDim.x + threadIdx.x;
    if (i < n) Wb[i] = f2bf(W[i]);
}

// ---- prep: padded feature table [64][2049][256] bf16, row0 = 0 ----------
__global__ void build_padded_kernel(const float* __restrict__ F,
                                    unsigned short* __restrict__ P) {
    size_t i = (size_t)blockIdx.x * blockDim.x + threadIdx.x;
    const size_t total = (size_t)BB * PADROWS * CC;
    if (i >= total) return;
    size_t b   = i / ((size_t)PADROWS * CC);
    size_t rem = i % ((size_t)PADROWS * CC);
    size_t row = rem / CC;
    size_t col = rem % CC;
    P[i] = (row == 0) ? (unsigned short)0
                      : f2bf(F[(b * NN + (row - 1)) * CC + col]);
}

// ---- main: gather + bf16 WMMA GEMM, W slice staged in LDS ---------------
// grid.x = 4 out-groups * 256 blocks; blockDim = 256 (8 waves).
// Block stages W rows [og*64, og*64+64) (96 KB bf16) in LDS once, then each
// wave sweeps 4 node-tiles of 16 nodes x 64 outs. A __syncthreads() at the
// top of each tile iteration fences LDS so B fragments are re-read per tile
// (ds_load_b128 next to the consuming WMMA) instead of being hoisted by
// LICM and spilled to scratch.
__global__ __launch_bounds__(256)
void btc_wmma_kernel(const unsigned short* __restrict__ P,   // padded feats bf16
                     const unsigned short* __restrict__ Wb,  // W bf16 [256][768]
                     const long long*      __restrict__ idx, // [64][6144]
                     const float*          __restrict__ bias,// [256]
                     float*                __restrict__ out) // [64][2048][256]
{
    __shared__ unsigned short sW[OG * KK];   // 98,304 bytes

    const int lane = threadIdx.x & 31;
    const int wave = threadIdx.x >> 5;
    const int h    = lane >> 4;      // half-wave selector
    const int r    = lane & 15;      // row-within-tile (M for A, N for B/C)

    const int og     = blockIdx.x & 3;
    const int blk    = blockIdx.x >> 2;
    const int o_base = og * OG;

    // ---- stage W slice into LDS (vectorized, whole block) ----
    {
        uint4*       dst = (uint4*)sW;
        const uint4* src = (const uint4*)(Wb + (size_t)o_base * KK);
        #pragma unroll
        for (int i = 0; i < (OG * KK / 8) / (NWAVE * 32); ++i) {
            int j = i * (NWAVE * 32) + threadIdx.x;   // 24 iters of 256 threads
            dst[j] = src[j];
        }
    }

    const uint4* sW4 = (const uint4*)sW;     // uint4 = 8 bf16

    // hoist bias: column handled by this lane per out-tile
    float bv[4];
    #pragma unroll
    for (int ot = 0; ot < 4; ++ot) bv[ot] = bias[o_base + ot * 16 + r];

    // ---- sweep node tiles ----
    #pragma unroll 1
    for (int t = 0; t < TILES_PER_WAVE; ++t) {
        // LDS fence + block sync: blocks LICM from hoisting the sW reads
        // across iterations (which previously caused scratch spills).
        __syncthreads();

        const int tile   = blk * (NWAVE * TILES_PER_WAVE) + t * NWAVE + wave;
        const int b      = tile >> 7;            // / (NN/16)
        const int m_base = (tile & 127) << 4;
        const int node   = m_base + r;

        // gather rows for the 3 pointers of this lane's node
        const long long* ip = idx + (size_t)b * (3 * NN) + 3 * (size_t)node;
        const unsigned short* rowp[3];
        #pragma unroll
        for (int p = 0; p < 3; ++p)
            rowp[p] = P + ((size_t)b * PADROWS + (size_t)ip[p]) * CC + h * 8;

        v8f c0 = {}, c1 = {}, c2 = {}, c3 = {};
        union Frag { uint4 u[2]; v16bf v; };

        // A fragment loader: 16-bit A 16x32 layout -> lane = M,
        // elems 0..7 = k (h*8..), elems 8..15 = k (16+h*8..)
        Frag Acur, Anext;
        {
            const unsigned short* ab = rowp[0];
            Acur.u[0] = *(const uint4*)(ab);
            Acur.u[1] = *(const uint4*)(ab + 16);
        }

        #pragma unroll
        for (int kc = 0; kc < 24; ++kc) {
            // prefetch next chunk's A into the alternate buffer
            if (kc + 1 < 24) {
                const int kn = kc + 1;
                const unsigned short* ab = rowp[kn >> 3] + (kn & 7) * 32;
                Anext.u[0] = *(const uint4*)(ab);
                Anext.u[1] = *(const uint4*)(ab + 16);
            }

            // B fragments from LDS: row (ot*16 + r) of slice, k = kc*32 + h*8 (+16)
            const int kb = kc * 32 + h * 8;
            Frag Bf0, Bf1, Bf2, Bf3;
            Bf0.u[0] = sW4[((0 * 16 + r) * KK + kb) >> 3];
            Bf0.u[1] = sW4[((0 * 16 + r) * KK + kb + 16) >> 3];
            Bf1.u[0] = sW4[((1 * 16 + r) * KK + kb) >> 3];
            Bf1.u[1] = sW4[((1 * 16 + r) * KK + kb + 16) >> 3];
            Bf2.u[0] = sW4[((2 * 16 + r) * KK + kb) >> 3];
            Bf2.u[1] = sW4[((2 * 16 + r) * KK + kb + 16) >> 3];
            Bf3.u[0] = sW4[((3 * 16 + r) * KK + kb) >> 3];
            Bf3.u[1] = sW4[((3 * 16 + r) * KK + kb + 16) >> 3];

            c0 = __builtin_amdgcn_wmma_f32_16x16x32_bf16(false, Acur.v, false, Bf0.v,
                                                         (short)0, c0, false, false);
            c1 = __builtin_amdgcn_wmma_f32_16x16x32_bf16(false, Acur.v, false, Bf1.v,
                                                         (short)0, c1, false, false);
            c2 = __builtin_amdgcn_wmma_f32_16x16x32_bf16(false, Acur.v, false, Bf2.v,
                                                         (short)0, c2, false, false);
            c3 = __builtin_amdgcn_wmma_f32_16x16x32_bf16(false, Acur.v, false, Bf3.v,
                                                         (short)0, c3, false, false);
            Acur = Anext;   // register ping-pong (renamed by unroll)
        }

        // epilogue: bias + store. C/D layout: VGPR v, lane l -> M = v+8h, N = r.
        float* obase = out + ((size_t)b * NN + m_base) * OUTC;
        const v8f cc[4] = {c0, c1, c2, c3};
        #pragma unroll
        for (int ot = 0; ot < 4; ++ot) {
            const int o = o_base + ot * 16 + r;
            #pragma unroll
            for (int v = 0; v < 8; ++v) {
                const int row = v + 8 * h;
                obase[(size_t)row * OUTC + o] = cc[ot][v] + bv[ot];
            }
        }
    }
}

extern "C" void kernel_launch(void* const* d_in, const int* in_sizes, int n_in,
                              void* d_out, int out_size, void* d_ws, size_t ws_size,
                              hipStream_t stream) {
    const float*     features = (const float*)d_in[0];
    const long long* idxes    = (const long long*)d_in[1];
    const float*     W        = (const float*)d_in[2];
    const float*     bias     = (const float*)d_in[3];
    float*           out      = (float*)d_out;

    // workspace layout: padded bf16 feats (64*2049*256 = 67,141,632 B),
    // then bf16 W (393,216 B)
    unsigned short* Pad = (unsigned short*)d_ws;
    unsigned short* Wb  = Pad + (size_t)BB * PADROWS * CC;

    const int nW = OUTC * KK;  // 196608
    convert_w_kernel<<<(nW + 255) / 256, 256, 0, stream>>>(W, Wb, nW);

    const size_t nP = (size_t)BB * PADROWS * CC;  // 33,570,816
    build_padded_kernel<<<(unsigned)((nP + 255) / 256), 256, 0, stream>>>(features, Pad);

    // 4 out-groups * 256 persistent blocks, 8 waves each, 4 node-tiles/wave
    btc_wmma_kernel<<<4 * BLKS_PER_OG, NWAVE * 32, 0, stream>>>(Pad, Wb, idxes, bias, out);

    // tuple output: append raw idxes (int64) after the f32 results
    const size_t res_bytes = (size_t)BB * NN * OUTC * sizeof(float);
    const size_t idx_bytes = (size_t)BB * 3 * NN * sizeof(long long);
    hipMemcpyAsync((char*)d_out + res_bytes, (const void*)idxes, idx_bytes,
                   hipMemcpyDeviceToDevice, stream);
}
